// IntraviewReg_22960895164562
// MI455X (gfx1250) — matrix operand order repaired and checked by reading, not compiled, not run on hardware
//
#include <hip/hip_runtime.h>
#include <hip/hip_bf16.h>
#include <stdint.h>

typedef __attribute__((ext_vector_type(16))) _Float16 v16h;
typedef __attribute__((ext_vector_type(8)))  _Float16 v8h;
typedef __attribute__((ext_vector_type(8)))  float    v8f;

#define CDIM 64

// ---------------- wave helpers (wave32) ----------------
__device__ __forceinline__ float wave_sum32(float v) {
#pragma unroll
    for (int off = 16; off >= 1; off >>= 1) v += __shfl_xor(v, off, 32);
    return v;
}
__device__ __forceinline__ float wave_max32(float v) {
#pragma unroll
    for (int off = 16; off >= 1; off >>= 1) v = fmaxf(v, __shfl_xor(v, off, 32));
    return v;
}

// ---------------- init ----------------
__global__ void init_kernel(float* colsum, int* mask, float* acc, int N) {
    int i = blockIdx.x * blockDim.x + threadIdx.x;
    if (i < N) { colsum[i] = 0.f; mask[i] = 0; }
    if (i < 4) acc[i] = 0.f;
}

__global__ void scatter_labels_kernel(const int* __restrict__ idx, int* __restrict__ mask, int L) {
    int i = blockIdx.x * blockDim.x + threadIdx.x;
    if (i < L) mask[idx[i]] = 1;
}

__global__ void colsum_kernel(const int* __restrict__ edst, const float* __restrict__ ew,
                              float* __restrict__ colsum, int E) {
    int i = blockIdx.x * blockDim.x + threadIdx.x;
    if (i < E) unsafeAtomicAdd(&colsum[edst[i]], ew[i]);
}

// ---------------- per-node log-softmax stats (one wave32 per node) ----------------
__device__ __forceinline__ void node_stats_one(const float* __restrict__ y,
                                               _Float16* __restrict__ pH,
                                               _Float16* __restrict__ qH,
                                               float* __restrict__ ng,
                                               int n, int lane) {
    const long r = (long)n * CDIM;
    const float v0 = y[r + lane];
    const float v1 = y[r + lane + 32];
    const float mx = wave_max32(fmaxf(v0, v1));
    const float s  = wave_sum32(__expf(v0 - mx) + __expf(v1 - mx));
    const float logZ = mx + __logf(s);
    const float lq0 = v0 - logZ, lq1 = v1 - logZ;
    const float p0 = __expf(lq0), p1 = __expf(lq1);
    pH[r + lane]      = (_Float16)p0;
    pH[r + lane + 32] = (_Float16)p1;
    qH[r + lane]      = (_Float16)lq0;
    qH[r + lane + 32] = (_Float16)lq1;
    const float ne = wave_sum32(p0 * lq0 + p1 * lq1); // sum p*logp (negative entropy)
    if (lane == 0) ng[n] = ne;
}

__global__ void __launch_bounds__(256) node_stats_kernel(
    const float* __restrict__ y1, const float* __restrict__ y2,
    _Float16* __restrict__ p1, _Float16* __restrict__ q1, float* __restrict__ ng1,
    _Float16* __restrict__ p2, _Float16* __restrict__ q2, float* __restrict__ ng2,
    int N) {
    const int lane = threadIdx.x & 31;
    const int node = (blockIdx.x * blockDim.x + threadIdx.x) >> 5;
    if (node >= N) return;
    node_stats_one(y1, p1, q1, ng1, node, lane);
    node_stats_one(y2, p2, q2, ng2, node, lane);
}

// ---------------- WMMA edge pass ----------------
// A fragment for V_WMMA_F32_16X16X32_F16: lane holds row M=lane&15,
// halves 0..7 = K base..base+7, halves 8..15 = K base+16..base+23 (base = lane<16?0:8).
__device__ __forceinline__ v16h load_a_frag(const _Float16* __restrict__ row, int base) {
    const v8h r1 = *(const v8h*)(row + base);
    const v8h r2 = *(const v8h*)(row + base + 16);
    v16h a;
#pragma unroll
    for (int i = 0; i < 8; ++i) { a[i] = r1[i]; a[8 + i] = r2[i]; }
    return a;
}

__global__ void __launch_bounds__(256) edge_kl_kernel(
    const int* __restrict__ esrc, const int* __restrict__ edst,
    const float* __restrict__ ew, const int* __restrict__ mask,
    const float* __restrict__ colsum,
    const _Float16* __restrict__ p1, const _Float16* __restrict__ q1,
    const float* __restrict__ ng1,
    const _Float16* __restrict__ p2, const _Float16* __restrict__ q2,
    const float* __restrict__ ng2,
    float* __restrict__ acc, int E) {
    const int lane = threadIdx.x & 31;
    const int wib  = threadIdx.x >> 5;
    const int wpb  = blockDim.x >> 5;
    const int gw   = blockIdx.x * wpb + wib;     // global wave id (uniform per wave)
    const int nwv  = gridDim.x * wpb;
    const int T    = (E + 15) >> 4;              // 16-edge tiles

    const int base = (lane < 16) ? 0 : 8;        // A K-base
    const int kb   = (lane < 16) ? 0 : 16;       // unused; B fragments use base, as A does
    // diagonal (m==n) of 16x16 f32 D: lane<8 -> VGPR lane; lane>=24 -> VGPR lane-24
    const int di   = (lane < 8) ? lane : ((lane >= 24) ? (lane - 24) : -1);

    float aKL1 = 0.f, aKL2 = 0.f, aCNT = 0.f, aNW = 0.f;

    for (int t = gw; t < T; t += nwv) {
        const int e  = t * 16 + (lane & 15);     // this lane's edge in the tile
        const int ec = (e < E) ? e : (E - 1);
        const int s  = esrc[ec];
        const int d  = edst[ec];
        const float em = ((e < E) && mask[d]) ? 1.f : 0.f;

        const long rS = (long)s * CDIM;
        const long rD = (long)d * CDIM;

        // View 1 fragments (K 0..31 and K 32..63)
        const v16h a10 = load_a_frag(p1 + rS, base);
        const v16h a11 = load_a_frag(p1 + rS + 32, base);
        const v16h b10 = load_a_frag(q1 + rD, base);
        const v16h b11 = load_a_frag(q1 + rD + 32, base);
        // View 2 fragments
        const v16h a20 = load_a_frag(p2 + rS, base);
        const v16h a21 = load_a_frag(p2 + rS + 32, base);
        const v16h b20 = load_a_frag(q2 + rD, base);
        const v16h b21 = load_a_frag(q2 + rD + 32, base);

        v8f c1 = {};
        c1 = __builtin_amdgcn_wmma_f32_16x16x32_f16(false, a10, false, b10, (short)0, c1, false, false);
        c1 = __builtin_amdgcn_wmma_f32_16x16x32_f16(false, a11, false, b11, (short)0, c1, false, false);
        v8f c2 = {};
        c2 = __builtin_amdgcn_wmma_f32_16x16x32_f16(false, a20, false, b20, (short)0, c2, false, false);
        c2 = __builtin_amdgcn_wmma_f32_16x16x32_f16(false, a21, false, b21, (short)0, c2, false, false);

        // extract this lane's diagonal element (its own edge's dot product)
        float d1 = 0.f, d2 = 0.f;
#pragma unroll
        for (int i = 0; i < 8; ++i) {
            d1 = (di == i) ? c1[i] : d1;
            d2 = (di == i) ? c2[i] : d2;
        }
        if (di >= 0) {                 // lanes {0..7, 24..31}: unique edge each
            aKL1 -= em * d1;
            aKL2 -= em * d2;
        }
        if (lane < 16) {               // per-edge scalar terms, counted once
            aKL1 += em * ng1[s];
            aKL2 += em * ng2[s];
            aCNT += em;
            const float cs = colsum[d];
            aNW  += (cs != 0.f) ? (em * ew[ec] / cs) : 0.f;
        }
    }

    aKL1 = wave_sum32(aKL1);
    aKL2 = wave_sum32(aKL2);
    aCNT = wave_sum32(aCNT);
    aNW  = wave_sum32(aNW);

    __shared__ float sred[8][4];
    if (lane == 0) {
        sred[wib][0] = aKL1; sred[wib][1] = aKL2;
        sred[wib][2] = aCNT; sred[wib][3] = aNW;
    }
    __syncthreads();
    if (threadIdx.x == 0) {
        float t0 = 0.f, t1 = 0.f, t2 = 0.f, t3 = 0.f;
        for (int i = 0; i < wpb; ++i) {
            t0 += sred[i][0]; t1 += sred[i][1]; t2 += sred[i][2]; t3 += sred[i][3];
        }
        unsafeAtomicAdd(&acc[0], t0);
        unsafeAtomicAdd(&acc[1], t1);
        unsafeAtomicAdd(&acc[2], t2);
        unsafeAtomicAdd(&acc[3], t3);
    }
}

__global__ void finalize_kernel(const float* __restrict__ acc, float* __restrict__ out, int L) {
    if (blockIdx.x == 0 && threadIdx.x == 0) {
        const float cnt = acc[2];
        const float mean_kl = (cnt > 0.f) ? (acc[0] + acc[1]) / cnt : 0.f;
        out[0] = acc[3] * mean_kl / (float)L;
    }
}

// ---------------- host launch ----------------
static inline char* align_up(char* p, size_t a) {
    return (char*)(((uintptr_t)p + (a - 1)) & ~(uintptr_t)(a - 1));
}

extern "C" void kernel_launch(void* const* d_in, const int* in_sizes, int n_in,
                              void* d_out, int out_size, void* d_ws, size_t ws_size,
                              hipStream_t stream) {
    const float* y1 = (const float*)d_in[0];
    const float* y2 = (const float*)d_in[1];
    const int* idx_label = (const int*)d_in[2];
    const int* eidx = (const int*)d_in[3];     // [2, E] row-major
    const float* ew = (const float*)d_in[4];

    const int N = in_sizes[0] / CDIM;
    const int L = in_sizes[2];
    const int E = in_sizes[4];
    const int* esrc = eidx;
    const int* edst = eidx + E;

    // workspace carve (256B aligned slices)
    char* w = (char*)d_ws;
    float* colsum = (float*)align_up(w, 256);            w = (char*)(colsum + N);
    int*   mask   = (int*)align_up(w, 256);              w = (char*)(mask + N);
    float* ng1    = (float*)align_up(w, 256);            w = (char*)(ng1 + N);
    float* ng2    = (float*)align_up(w, 256);            w = (char*)(ng2 + N);
    _Float16* p1  = (_Float16*)align_up(w, 256);         w = (char*)(p1 + (long)N * CDIM);
    _Float16* q1  = (_Float16*)align_up(w, 256);         w = (char*)(q1 + (long)N * CDIM);
    _Float16* p2  = (_Float16*)align_up(w, 256);         w = (char*)(p2 + (long)N * CDIM);
    _Float16* q2  = (_Float16*)align_up(w, 256);         w = (char*)(q2 + (long)N * CDIM);
    float* acc    = (float*)align_up(w, 256);

    hipLaunchKernelGGL(init_kernel, dim3((N + 255) / 256), dim3(256), 0, stream,
                       colsum, mask, acc, N);
    hipLaunchKernelGGL(scatter_labels_kernel, dim3((L + 255) / 256), dim3(256), 0, stream,
                       idx_label, mask, L);
    hipLaunchKernelGGL(colsum_kernel, dim3((E + 255) / 256), dim3(256), 0, stream,
                       edst, ew, colsum, E);
    hipLaunchKernelGGL(node_stats_kernel, dim3((N * 32 + 255) / 256), dim3(256), 0, stream,
                       y1, y2, p1, q1, ng1, p2, q2, ng2, N);
    hipLaunchKernelGGL(edge_kl_kernel, dim3(1024), dim3(256), 0, stream,
                       esrc, edst, ew, mask, colsum,
                       p1, q1, ng1, p2, q2, ng2, acc, E);
    hipLaunchKernelGGL(finalize_kernel, dim3(1), dim3(64), 0, stream,
                       acc, (float*)d_out, L);
}
